// ConvCapsNet1D_28338194219365
// MI455X (gfx1250) — matrix operand order repaired and checked by reading, not compiled, not run on hardware
//
#include <hip/hip_runtime.h>
#include <hip/hip_bf16.h>
#include <math.h>

// Problem constants (from reference)
#define B_SZ   16
#define IC     16
#define LEN    4096
#define OC     256
#define NCAPS  8
#define CD     16
#define KW     9
#define M_DIM  128          // CD*NCAPS rows of fused weight
#define K_DIM  160          // IC*KW = 144, padded to 5 chunks of 32
#define KC     5
#define S_PER_WG 128
#define XPITCH 24           // ushorts per LDS row (16 data + 8 pad) = 48B, bank-conflict free
#define XROWS  140
#define AW_USHORTS (8 * KC * 512)   // 20480 ushorts = 40960 B fragment table

typedef __attribute__((ext_vector_type(16))) __bf16          v16bf;
typedef __attribute__((ext_vector_type(8)))  float           v8f;
typedef __attribute__((ext_vector_type(8)))  unsigned short  u16x8;

union Frag16 { v16bf v; u16x8 h[2]; unsigned short u[16]; };
union Acc8   { v8f   v; float f[8]; };

__device__ __forceinline__ unsigned short f32_to_bf16(float f) {
    union { float f; unsigned int u; } x; x.f = f;
    unsigned int r = x.u + 0x7FFFu + ((x.u >> 16) & 1u);  // round-to-nearest-even
    return (unsigned short)(r >> 16);
}

// gfx1250 async DMA: copy 16B from global directly into LDS, tracked by ASYNCcnt
// (GLOBAL_LOAD_ASYNC_TO_LDS_B128, CDNA5 ISA; GV addressing mode).
__device__ __forceinline__ void async_copy_b128(unsigned lds_byte_off, const void* g) {
    asm volatile("global_load_async_to_lds_b128 %0, %1, off"
                 :: "v"(lds_byte_off), "v"(g)
                 : "memory");
}
__device__ __forceinline__ void wait_asynccnt0() {
    asm volatile("s_wait_asynccnt 0" ::: "memory");
}

// ---------------------------------------------------------------------------
// Kernel A1: c = softmax(W, axis=IC)  (constant across routing iterations),
// and combined bias cb[m] = caps_b[n,k] + caps_w[n,k,:]·conv_b
// ---------------------------------------------------------------------------
__global__ void capsnet_prep_small(const float* __restrict__ conv_b,
                                   const float* __restrict__ caps_w,
                                   const float* __restrict__ caps_b,
                                   const float* __restrict__ Wroute,
                                   float* __restrict__ c_ws,
                                   float* __restrict__ cb_ws) {
    int t = threadIdx.x;            // 0..127
    int n = t >> 4;                 // caps / n_out index
    int k = t & 15;                 // CD index
    float w[16];
    float mx = -1e30f;
    for (int b = 0; b < 16; ++b) {
        w[b] = Wroute[(n * 16 + b) * 16 + k];
        mx = fmaxf(mx, w[b]);
    }
    float sum = 0.f;
    for (int b = 0; b < 16; ++b) { w[b] = __expf(w[b] - mx); sum += w[b]; }
    float inv = 1.0f / sum;
    for (int b = 0; b < 16; ++b)
        c_ws[(k * 8 + n) * 16 + b] = w[b] * inv;   // layout [k][n_out][b]

    float acc = caps_b[n * 16 + k];
    for (int o = 0; o < OC; ++o)
        acc += caps_w[(n * 16 + k) * OC + o] * conv_b[o];
    cb_ws[k * 8 + n] = acc;                         // m = k*8 + n
}

// ---------------------------------------------------------------------------
// Kernel A2: fused weight cw[m][kidx] = sum_o caps_w[n,k,o]*conv_w[o,c,kk],
// stored PRE-SWIZZLED as WMMA A-fragments (contiguous-K order used here, not the ISA table order):
// lane half h, element j:  k = 16h + j
// cwf[(mt*KC+kc)*512 + lane*16 + j], bf16.  kidx >= 144 zero-padded.
// ---------------------------------------------------------------------------
__global__ void capsnet_fuse_weights(const float* __restrict__ conv_w,
                                     const float* __restrict__ caps_w,
                                     unsigned short* __restrict__ cwf) {
    int f  = blockIdx.x;                 // 0..39 : f = mt*KC + kc
    int mt = f / KC, kc = f % KC;
    int t  = threadIdx.x;                // 0..511
    int lane = t >> 4, j = t & 15;
    int h = lane >> 4, lm = lane & 15;
    int m = mt * 16 + lm;
    int v = j >> 1, r = j & 1;
    int kfrag = 16 * h + j;
    int kidx = kc * 32 + kfrag;
    float acc = 0.f;
    if (kidx < IC * KW) {                // 144 real K values
        int kk = kidx >> 4, cc = kidx & 15;    // kidx = kk*16 + c
        int kcd = m >> 3, n = m & 7;           // m = kcd*8 + n
        const float* cwp = caps_w + (n * 16 + kcd) * OC;
        for (int o = 0; o < OC; ++o)
            acc += cwp[o] * conv_w[(o * IC + cc) * KW + kk];
    }
    cwf[f * 512 + lane * 16 + j] = f32_to_bf16(acc);
}

// ---------------------------------------------------------------------------
// Kernel B: implicit-GEMM conv+capsule. Per WG: one (b, 128-column s chunk).
//  - 40 KB fused-weight fragment table DMA'd into LDS via
//    global_load_async_to_lds_b128 (ASYNCcnt), so the K-loop A operand is
//    pure ds_load_b128 instead of latency-bound global loads.
//  - x staged in LDS as bf16 [s_off][c] rows (48B pitch, conflict-free).
//  - Each wave owns a 16-column s-tile; B fragment per K-chunk is two
//    ds_load_b128 (B 32x16 blocked-K layout: half h -> K=16h..16h+15).
//  - M loop unrolled x2 -> two independent WMMA accumulate chains.
// u stored [m][b][s] for coalesced routing reads.
// ---------------------------------------------------------------------------
__global__ __launch_bounds__(256) void capsnet_conv_gemm(
        const float* __restrict__ x,
        const unsigned short* __restrict__ cwf,
        const float* __restrict__ cb_ws,
        float* __restrict__ u_ws) {
    __shared__ unsigned short xlds[XROWS * XPITCH];
    __shared__ unsigned short awlds[AW_USHORTS];
    const int b      = blockIdx.x >> 5;          // 16 batches
    const int chunk  = blockIdx.x & 31;          // 32 chunks of 128 columns
    const int s_base = chunk * S_PER_WG;
    const int tid    = threadIdx.x;

    // Kick off the async DMA of the A-fragment table (2560 x 16B, 10/lane)
    const unsigned aw_base = (unsigned)(unsigned long long)(void*)awlds;
    #pragma unroll
    for (int it = 0; it < 10; ++it) {
        int i = tid + 256 * it;
        async_copy_b128(aw_base + (unsigned)i * 16u,
                        (const char*)cwf + (size_t)i * 16u);
    }

    if (tid < 16)                                 // warm next x rows in L2
        __builtin_prefetch(&x[(b * IC + tid) * LEN + s_base], 0, 1);

    // Stage x[b][c][s_base-4 .. s_base+132] -> xlds[ri][c] (zero conv halo)
    for (int idx = tid; idx < 137 * 16; idx += 256) {
        int ri = idx >> 4, c = idx & 15;
        int gs = s_base - 4 + ri;
        float val = (gs >= 0 && gs < LEN) ? x[(b * IC + c) * LEN + gs] : 0.f;
        xlds[ri * XPITCH + c] = f32_to_bf16(val);
    }
    wait_asynccnt0();          // this wave's DMA into LDS complete
    __syncthreads();           // all waves' staging visible

    const int wave = tid >> 5;
    const int lane = tid & 31;
    const int n    = lane & 15;
    const int h    = lane >> 4;
    const int s    = s_base + wave * 16 + n;

    // B fragments: kidx = kc*32 + 16h + j  ->  kk = 2kc+h, c = j
    // row ri = (s - s_base) + kk; one 16-wide c-row = two b128 LDS reads.
    Frag16 bfr[KC];
    #pragma unroll
    for (int kc = 0; kc < KC; ++kc) {
        int ri = wave * 16 + n + 2 * kc + h;
        const unsigned short* row = &xlds[ri * XPITCH];
        bfr[kc].h[0] = *(const u16x8*)(row);
        bfr[kc].h[1] = *(const u16x8*)(row + 8);
    }

    #pragma unroll 2
    for (int mt = 0; mt < 8; ++mt) {
        Acc8 acc;
        #pragma unroll
        for (int i = 0; i < 8; ++i) acc.f[i] = 0.f;
        #pragma unroll
        for (int kc = 0; kc < KC; ++kc) {
            Frag16 af;
            const unsigned short* ap = &awlds[(mt * KC + kc) * 512 + lane * 16];
            af.h[0] = *(const u16x8*)(ap);
            af.h[1] = *(const u16x8*)(ap + 8);
            acc.v = __builtin_amdgcn_wmma_f32_16x16x32_bf16(
                false, af.v, false, bfr[kc].v, (short)0, acc.v, false, false);
        }
        // C/D layout: lane: N = lane%16, VGPR v: M = v + 8*(lane/16)
        #pragma unroll
        for (int vv = 0; vv < 8; ++vv) {
            int m = mt * 16 + vv + 8 * h;
            u_ws[((size_t)m * B_SZ + b) * LEN + s] = acc.f[vv] + cb_ws[m];
        }
    }
}

// ---------------------------------------------------------------------------
// Kernel C: routing. Because softmax(W) sums to 1 over IC, the 3 iterations
// reduce to (per n_out, per m=(k,n), elementwise in s):
//   s0 = sum_b c[n_out,b,k]*u[b],  U = sum_b u[b]
//   a1 = alpha(s0);  s1 = s0 + a1*s0^2*U
//   a2 = alpha(s1);  s2 = s0 + a2*s1^2*U
//   out = alpha(s2) * s2,   alpha(s) = ||s||/(1+||s||^2), norm over S=4096.
// One block per m; thread-register tiles of 16 s-values; u is L2-resident.
// ---------------------------------------------------------------------------
__device__ __forceinline__ float block_sum(float v, float* red, int tid) {
    __syncthreads();
    red[tid] = v;
    __syncthreads();
    for (int off = 128; off > 0; off >>= 1) {
        if (tid < off) red[tid] += red[tid + off];
        __syncthreads();
    }
    return red[0];
}

__global__ __launch_bounds__(256) void capsnet_route(
        const float* __restrict__ u_ws,
        const float* __restrict__ c_ws,
        float* __restrict__ out) {
    __shared__ float red[256];
    const int p    = blockIdx.x;        // m = kcd*8 + ncap
    const int kcd  = p >> 3;
    const int ncap = p & 7;
    const int tid  = threadIdx.x;

    float U[16];
    #pragma unroll
    for (int j = 0; j < 16; ++j) U[j] = 0.f;
    for (int b = 0; b < 16; ++b) {
        const float* up = u_ws + ((size_t)p * 16 + b) * LEN;
        #pragma unroll
        for (int j = 0; j < 16; ++j) U[j] += up[tid + 256 * j];
    }

    for (int nout = 0; nout < 8; ++nout) {
        float s0[16];
        #pragma unroll
        for (int j = 0; j < 16; ++j) s0[j] = 0.f;
        for (int b = 0; b < 16; ++b) {
            float w = c_ws[(kcd * 8 + nout) * 16 + b];
            const float* up = u_ws + ((size_t)p * 16 + b) * LEN;
            #pragma unroll
            for (int j = 0; j < 16; ++j) s0[j] += w * up[tid + 256 * j];
        }
        float loc = 0.f;
        #pragma unroll
        for (int j = 0; j < 16; ++j) loc += s0[j] * s0[j];
        float n2 = block_sum(loc, red, tid);
        float a1 = sqrtf(n2) / (1.0f + n2);

        float s1[16];
        loc = 0.f;
        #pragma unroll
        for (int j = 0; j < 16; ++j) {
            s1[j] = s0[j] + a1 * s0[j] * s0[j] * U[j];
            loc += s1[j] * s1[j];
        }
        n2 = block_sum(loc, red, tid);
        float a2 = sqrtf(n2) / (1.0f + n2);

        float s2[16];
        loc = 0.f;
        #pragma unroll
        for (int j = 0; j < 16; ++j) {
            s2[j] = s0[j] + a2 * s1[j] * s1[j] * U[j];
            loc += s2[j] * s2[j];
        }
        n2 = block_sum(loc, red, tid);
        float a3 = sqrtf(n2) / (1.0f + n2);

        float* op = out + ((size_t)(nout * 16 + kcd) * 8 + ncap) * LEN;
        #pragma unroll
        for (int j = 0; j < 16; ++j) op[tid + 256 * j] = a3 * s2[j];
    }
}

// ---------------------------------------------------------------------------
extern "C" void kernel_launch(void* const* d_in, const int* in_sizes, int n_in,
                              void* d_out, int out_size, void* d_ws, size_t ws_size,
                              hipStream_t stream) {
    (void)in_sizes; (void)n_in; (void)out_size; (void)ws_size;
    const float* x      = (const float*)d_in[0];
    const float* conv_w = (const float*)d_in[1];
    const float* conv_b = (const float*)d_in[2];
    const float* caps_w = (const float*)d_in[3];
    const float* caps_b = (const float*)d_in[4];
    const float* Wrt    = (const float*)d_in[5];
    float* out = (float*)d_out;

    char* ws = (char*)d_ws;
    float*          c_ws  = (float*)(ws + 0);          //  8 KB  [k][n_out][b]
    float*          cb_ws = (float*)(ws + 8192);       // 512 B  combined bias
    unsigned short* cwf   = (unsigned short*)(ws + 8704); // 40 KB pre-swizzled A frags
    float*          u_ws  = (float*)(ws + 65536);      // 32 MB  u[m][b][s]

    capsnet_prep_small <<<1,        128, 0, stream>>>(conv_b, caps_w, caps_b, Wrt, c_ws, cb_ws);
    capsnet_fuse_weights<<<8 * KC,  512, 0, stream>>>(conv_w, caps_w, cwf);
    capsnet_conv_gemm  <<<16 * 32,  256, 0, stream>>>(x, cwf, cb_ws, u_ws);
    capsnet_route      <<<M_DIM,    256, 0, stream>>>(u_ws, c_ws, out);
}
